// ICS_38714835206340
// MI455X (gfx1250) — matrix operand order repaired and checked by reading, not compiled, not run on hardware
//
#include <hip/hip_runtime.h>
#include <hip/hip_bf16.h>
#include <math.h>
#include <stdint.h>

// ---------------------------------------------------------------------------
// Types for CDNA5 WMMA (wave32): D(f32 16x16) = A(bf16 16x32) * B(bf16 32x16) + C
// ---------------------------------------------------------------------------
typedef __attribute__((ext_vector_type(16))) __bf16 bf16x16;
typedef __attribute__((ext_vector_type(8)))  __bf16 bf16x8;
typedef __attribute__((ext_vector_type(8)))  float  f32x8;
typedef __attribute__((ext_vector_type(4)))  unsigned int u32x4;
typedef __attribute__((ext_vector_type(8)))  int    i32x8;
typedef __attribute__((ext_vector_type(4)))  int    i32x4;

#if __has_builtin(__builtin_amdgcn_tensor_load_to_lds) && \
    __has_builtin(__builtin_amdgcn_s_wait_tensorcnt)
#define HAVE_TDM 1
#endif

static __device__ __forceinline__ f32x8 wmma_bf16(bf16x16 a, bf16x16 b, f32x8 c) {
    // 8 args: (neg_a, A, neg_b, B, c_mod, C, reuse_a, reuse_b)
    return __builtin_amdgcn_wmma_f32_16x16x32_bf16(false, a, false, b, (short)0, c,
                                                   false, false);
}

// A-fragment (16x32 bf16, ISA 7.12.2): lane l holds row M=(l&15); lanes 0-15
// carry K[k0..k0+7] / K[k0+16..k0+23] with k0=0, lanes 16-31 with k0=8.
static __device__ __forceinline__ bf16x16 load_fragA(const __bf16* row, int k0) {
    bf16x8 lo = *(const bf16x8*)(row + k0);
    bf16x8 hi = *(const bf16x8*)(row + k0 + 16);
    return __builtin_shufflevector(lo, hi, 0,1,2,3,4,5,6,7,8,9,10,11,12,13,14,15);
}

// B-fragment (32x16 bf16): lane l holds column N=(l&15); lanes 0-15 carry
// K[k0..k0+7] / K[k0+16..k0+23] with k0=0, lanes 16-31 with k0=8.  Source stored K-contiguous per column.
static __device__ __forceinline__ bf16x16 load_fragB(const __bf16* colrow, int k0) {
    bf16x8 lo = *(const bf16x8*)(colrow + k0);
    bf16x8 hi = *(const bf16x8*)(colrow + k0 + 16);
    return __builtin_shufflevector(lo, hi, 0,1,2,3,4,5,6,7,8,9,10,11,12,13,14,15);
}

// ---------------------------------------------------------------------------
// Kernel 1/4: 1x1 conv as GEMM.  Y[b][m][p] = sum_k W[m][k] * X[b][k][p] + bias[m]
// ---------------------------------------------------------------------------
#define XT_STRIDE 104   // 96 + 8 pad, 16B multiple

__global__ __launch_bounds__(256)
void conv1x1_wmma_kernel(const float* __restrict__ X, const float* __restrict__ Wm,
                         const float* __restrict__ bias, float* __restrict__ Y,
                         int M, int K, int HW) {
    const int ptiles = HW >> 8;
    const int b  = blockIdx.x / ptiles;
    const int p0 = (blockIdx.x % ptiles) << 8;

    extern __shared__ char smem[];
    __bf16* wq = (__bf16*)smem;          // [M][K] row-major (K contiguous)
    __bf16* xT = wq + M * K;             // [256][XT_STRIDE]

    const int tid  = threadIdx.x;
    const int lane = tid & 31;
    const int wv   = tid >> 5;

    for (int i = tid; i < M * K; i += 256) wq[i] = (__bf16)Wm[i];
    const float* xb = X + (size_t)b * K * HW;
    for (int c = 0; c < K; ++c)
        xT[tid * XT_STRIDE + c] = (__bf16)xb[(size_t)c * HW + p0 + tid];
    __syncthreads();

    const int mtiles = M >> 4;
    const int T = mtiles * 16;
    const int ak  = (lane < 16) ? 0 : 8;
    const int bk  = (lane < 16) ? 0 : 16;
    const int hi8 = (lane < 16) ? 0 : 8;
    float* yb = Y + (size_t)b * M * HW;

    for (int t = wv; t < T; t += 8) {
        const int mt = t >> 4, nt = t & 15;
        const int mrow = mt * 16 + (lane & 15);
        const int ncol = nt * 16 + (lane & 15);
        f32x8 acc = {};
        for (int kk = 0; kk < K; kk += 32) {
            bf16x16 A = load_fragA(wq + mrow * K, kk + ak);
            bf16x16 B = load_fragB(xT + ncol * XT_STRIDE, kk + ak);
            acc = wmma_bf16(A, B, acc);
        }
        for (int r = 0; r < 8; ++r) {
            const int mr = mt * 16 + r + hi8;
            yb[(size_t)mr * HW + p0 + nt * 16 + (lane & 15)] = acc[r] + bias[mr];
        }
    }
}

// ---------------------------------------------------------------------------
// Kernel 2: depthwise 3x3 SAME.  q,k channels -> f32; v channels -> bf16 so the
// attention kernel can DMA them straight to LDS via the Tensor Data Mover.
// ---------------------------------------------------------------------------
__global__ __launch_bounds__(256)
void dwconv3x3_kernel(const float* __restrict__ in, const float* __restrict__ w9,
                      const float* __restrict__ bias,
                      float* __restrict__ qk_out, __bf16* __restrict__ v_out,
                      int BC3) {
    const size_t idx = (size_t)blockIdx.x * 256 + threadIdx.x;
    const int bc = (int)(idx >> 16);
    if (bc >= BC3) return;
    const int p = (int)(idx & 65535);
    const int h = p >> 8, x = p & 255;
    const int oc = bc % 288, b = bc / 288;
    const float* ip = in + ((size_t)bc << 16);
    const float* wc = w9 + oc * 9;
    float acc = bias[oc];
#pragma unroll
    for (int dy = -1; dy <= 1; ++dy) {
        const int hh = h + dy;
        if (hh < 0 || hh > 255) continue;
#pragma unroll
        for (int dx = -1; dx <= 1; ++dx) {
            const int xx = x + dx;
            if (xx < 0 || xx > 255) continue;
            acc += wc[(dy + 1) * 3 + (dx + 1)] * ip[hh * 256 + xx];
        }
    }
    if (oc < 192) qk_out[((size_t)(b * 192 + oc)       << 16) + p] = acc;
    else          v_out [((size_t)(b *  96 + oc - 192) << 16) + p] = (__bf16)acc;
}

// ---------------------------------------------------------------------------
// Kernel 3: fused per-channel attention, one WG (8 wave32) per (b,c).
// LDS: q1^T [256][264]bf16 132KB + V [256][264]bf16 132KB (TDM-loaded with
// per-row padding) + per-16-column block: k1^T 8.4KB, attn f32 16.9KB,
// P^T bf16 8.4KB, k-norms 1KB  => ~298KB of the 320KB WGP LDS.
// ---------------------------------------------------------------------------
#define QSTR 264                     // 256 + 8 pad (16B multiple in bf16)
#define VM_LDS_OFF (256 * QSTR * 2)  // byte offset of vm inside dynamic LDS

__global__ __launch_bounds__(256)
void attn_channel_kernel(const float* __restrict__ qk, const __bf16* __restrict__ vbf,
                         const float* __restrict__ temp, float* __restrict__ obuf) {
    const int bc = blockIdx.x;           // b*96 + c
    const int b = bc / 96, c = bc % 96;
    const float*  qd   = qk  + ((size_t)(b * 192 +      c) << 16);
    const float*  kd   = qk  + ((size_t)(b * 192 + 96 + c) << 16);
    const __bf16* vsrc = vbf + ((size_t)bc << 16);
    float* od = obuf + ((size_t)bc << 16);
    const float tval = temp[c];

    extern __shared__ char smem[];
    __bf16* qt = (__bf16*)smem;              // [256][QSTR]  q1^T
    __bf16* vm = qt + 256 * QSTR;            // [256][QSTR]  V (row h, w contiguous)
    __bf16* kt = vm + 256 * QSTR;            // [16][QSTR]   k1 block^T
    float*  at = (float*)(kt + 16 * QSTR);   // [16][QSTR]   attn^T f32
    __bf16* pt = (__bf16*)(at + 16 * QSTR);  // [16][QSTR]   softmax^T bf16
    float*  kn = (float*)(pt + 16 * QSTR);   // [256] k column inv-norms

    const int tid  = threadIdx.x;
    const int lane = tid & 31;
    const int wv   = tid >> 5;
    const int ak   = (lane < 16) ? 0 : 8;
    const int bk   = (lane < 16) ? 0 : 16;
    const int hi8  = (lane < 16) ? 0 : 8;
    const int nl   = lane & 15;

#ifdef HAVE_TDM
    // Kick off the V-matrix DMA (global bf16 [256][256] -> LDS [256][QSTR]) on
    // wave 0 so it overlaps with the q/k norm pass.  D# per ISA 8.3/8.4:
    //   g0: count=1 | lds_addr | global_addr | type=2
    //   g1: data_size=2B, pad_enable, pad_interval=128DW, pad_amount=4DW,
    //       tensor_dim0=256, tensor_dim1=256, tile_dim0=256, tile_dim1=256,
    //       tensor_dim0_stride=256
    // This toolchain exposes the 6-arg form: (g0, g1, g2, g3, g4, cpol);
    // unused groups are zero (tile is 2-D).
    if (wv == 0) {
        const unsigned long long ga = (unsigned long long)(uintptr_t)vsrc;
        u32x4 g0 = { 1u,                       // count=1, user descriptor
                     (unsigned)VM_LDS_OFF,
                     (unsigned)ga,
                     (unsigned)((ga >> 32) & 0x1FFFFFFu) | (2u << 30) };
        i32x8 g1 = { (int)((1u << 16) | (1u << 20) | (6u << 22) | (3u << 25)),
                     (int)(256u << 16),        // tensor_dim0[15:0]
                     (int)(256u << 16),        // tensor_dim1[15:0]
                     (int)(256u << 16),        // tile_dim0
                     (int)256,                 // tile_dim1
                     (int)256,                 // tensor_dim0_stride[31:0]
                     0, 0 };
        i32x4 gz4 = { 0, 0, 0, 0 };
        i32x8 gz8 = { 0, 0, 0, 0, 0, 0, 0, 0 };
        __builtin_amdgcn_tensor_load_to_lds(g0, g1, gz4, gz4, gz8, 0);
    }
#endif

    // Column L2 norms over H for q (col tid) and k; transpose-store q1.
    float sq = 0.f, sk = 0.f;
    for (int h = 0; h < 256; ++h) {
        const float qv = qd[h * 256 + tid]; sq += qv * qv;
        const float kv = kd[h * 256 + tid]; sk += kv * kv;
    }
    const float qs = 1.0f / fmaxf(sqrtf(sq), 1e-12f);
    kn[tid]        = 1.0f / fmaxf(sqrtf(sk), 1e-12f);
    for (int h = 0; h < 256; ++h)
        qt[tid * QSTR + h] = (__bf16)(qd[h * 256 + tid] * qs);

#ifdef HAVE_TDM
    if (wv == 0) __builtin_amdgcn_s_wait_tensorcnt(0);   // V tile landed in LDS
#else
    for (int h = 0; h < 256; ++h)
        vm[h * QSTR + tid] = vsrc[h * 256 + tid];
#endif
    __syncthreads();

    for (int vb = 0; vb < 16; ++vb) {
        const int v0 = vb * 16;
        // stream normalized k columns v0..v0+15 into LDS (transposed)
        {
            const int j = tid & 15, hh = tid >> 4;
            const float ks = kn[v0 + j];
            for (int hi = 0; hi < 16; ++hi) {
                const int h = hi * 16 + hh;
                kt[j * QSTR + h] = (__bf16)(kd[h * 256 + v0 + j] * ks);
            }
        }
        __syncthreads();

        // GEMM1: attn[w, v] = sum_h q1[h,w] k1[h,v]   (M=256, N=16, K=256)
        for (int mt = wv; mt < 16; mt += 8) {
            const int mrow = mt * 16 + nl;
            f32x8 acc = {};
#pragma unroll
            for (int kk = 0; kk < 8; ++kk) {
                bf16x16 A = load_fragA(qt + mrow * QSTR, kk * 32 + ak);
                bf16x16 B = load_fragB(kt + nl * QSTR,   kk * 32 + ak);
                acc = wmma_bf16(A, B, acc);
            }
#pragma unroll
            for (int r = 0; r < 8; ++r)
                at[nl * QSTR + mt * 16 + r + hi8] = acc[r] * tval;
        }
        __syncthreads();

        // softmax over w for each of the 16 columns; 16 threads per column
        {
            const int v = tid >> 4, j = tid & 15;
            float e[16];
            float mx = -INFINITY;
#pragma unroll
            for (int i = 0; i < 16; ++i) {
                e[i] = at[v * QSTR + j * 16 + i];
                mx = fmaxf(mx, e[i]);
            }
            for (int off = 8; off; off >>= 1) mx = fmaxf(mx, __shfl_xor(mx, off, 16));
            float s = 0.f;
#pragma unroll
            for (int i = 0; i < 16; ++i) { e[i] = __expf(e[i] - mx); s += e[i]; }
            for (int off = 8; off; off >>= 1) s += __shfl_xor(s, off, 16);
            const float inv = 1.0f / s;
#pragma unroll
            for (int i = 0; i < 16; ++i)
                pt[v * QSTR + j * 16 + i] = (__bf16)(e[i] * inv);
        }
        __syncthreads();

        // GEMM2: out[h, v] = sum_w V[h,w] P[w,v]   (M=256, N=16, K=256)
        for (int mt = wv; mt < 16; mt += 8) {
            const int mrow = mt * 16 + nl;
            f32x8 acc = {};
#pragma unroll
            for (int kk = 0; kk < 8; ++kk) {
                bf16x16 A = load_fragA(vm + mrow * QSTR, kk * 32 + ak);
                bf16x16 B = load_fragB(pt + nl * QSTR,   kk * 32 + ak);
                acc = wmma_bf16(A, B, acc);
            }
#pragma unroll
            for (int r = 0; r < 8; ++r)
                od[(mt * 16 + r + hi8) * 256 + v0 + nl] = acc[r];
        }
        __syncthreads();
    }
}

// ---------------------------------------------------------------------------
// Launcher.  Workspace: qkv_lin (302MB) | qk_f32 (201MB) | v_bf16 (50MB);
// attention output aliases qkv_lin (dead after dwconv).  Peak ~553MB.
// ---------------------------------------------------------------------------
extern "C" void kernel_launch(void* const* d_in, const int* in_sizes, int n_in,
                              void* d_out, int out_size, void* d_ws, size_t ws_size,
                              hipStream_t stream) {
    const float* x      = (const float*)d_in[0];
    const float* qkv_w  = (const float*)d_in[1];
    const float* qkv_b  = (const float*)d_in[2];
    const float* dw_w   = (const float*)d_in[3];
    const float* dw_b   = (const float*)d_in[4];
    const float* temp   = (const float*)d_in[5];
    const float* proj_w = (const float*)d_in[6];
    const float* proj_b = (const float*)d_in[7];
    float* out = (float*)d_out;

    const int HW = 256 * 256;
    const size_t qkvElems = (size_t)4 * 288 * HW;
    float*  qkv_lin = (float*)d_ws;
    float*  qk_f32  = qkv_lin + qkvElems;
    __bf16* v_bf16  = (__bf16*)(qk_f32 + (size_t)4 * 192 * HW);
    float*  obuf    = qkv_lin;           // alias: qkv_lin dead after dwconv

    // K1: QKV 1x1 conv  (M=288, K=96)
    conv1x1_wmma_kernel<<<4 * (HW / 256), 256, 288 * 96 * 2 + 256 * XT_STRIDE * 2,
                          stream>>>(x, qkv_w, qkv_b, qkv_lin, 288, 96, HW);

    // K2: depthwise 3x3 (q,k -> f32 ; v -> bf16 for TDM staging)
    dwconv3x3_kernel<<<(4 * 288 * HW) / 256, 256, 0, stream>>>(qkv_lin, dw_w, dw_b,
                                                               qk_f32, v_bf16, 4 * 288);

    // K3: fused attention, one WG per (b,c); LDS ~= 298KB
    const size_t attn_lds = (size_t)(2 * 256 * QSTR * 2)   // qt + vm
                          + 16 * QSTR * 2                  // kt
                          + 16 * QSTR * 4                  // at (f32)
                          + 16 * QSTR * 2                  // pt
                          + 256 * 4;                       // kn
    attn_channel_kernel<<<4 * 96, 256, attn_lds, stream>>>(qk_f32, v_bf16, temp, obuf);

    // K4: output 1x1 projection (M=96, K=96)
    conv1x1_wmma_kernel<<<4 * (HW / 256), 256, 96 * 96 * 2 + 256 * XT_STRIDE * 2,
                          stream>>>(obuf, proj_w, proj_b, out, 96, 96, HW);
}
